// AntecedentShareGMF_32263794327590
// MI455X (gfx1250) — compile-verified, matrix operand order checked by reading, not run on hardware
//
#include <hip/hip_runtime.h>
#include <hip/hip_bf16.h>

typedef __attribute__((ext_vector_type(16))) _Float16 v16h;
typedef __attribute__((ext_vector_type(8)))  float    v8f;

#define D_FEAT 10
#define M_MF   2
#define R_RULES 1024          // M^D
#define K_PAD  32             // K padded for 16x16x32 f16 WMMA
#define EPS_F  1e-8f

// ---------------------------------------------------------------------------
// Kernel 0: build WMMA operands in workspace.
//   Aws[n*32 + k]  : k<10 -> X[n,k]^2 ; 10<=k<20 -> X[n,k] ; else 0      (f16)
//   Bws[r*32 + k]  : k<10 -> W[r,k]   ; 10<=k<20 -> -2*C*W ; else 0      (f16)
//   bias[r]        : sum_d C^2 * W                                        (f32)
// where W = 1/(sigma^2 + eps), C/S selected by rule bit pattern of r.
// ---------------------------------------------------------------------------
__global__ void prep_kernel(const float* __restrict__ X,
                            const float* __restrict__ center,
                            const float* __restrict__ sigma,
                            _Float16* __restrict__ Aws,
                            _Float16* __restrict__ Bws,
                            float* __restrict__ biasws,
                            int N) {
  int tid = blockIdx.x * blockDim.x + threadIdx.x;

  if (tid < N) {
    const float* xr = X + tid * D_FEAT;
    _Float16* arow = Aws + tid * K_PAD;
#pragma unroll
    for (int d = 0; d < D_FEAT; ++d) {
      float x = xr[d];
      arow[d]          = (_Float16)(x * x);
      arow[D_FEAT + d] = (_Float16)x;
    }
#pragma unroll
    for (int k = 2 * D_FEAT; k < K_PAD; ++k) arow[k] = (_Float16)0.0f;
  }

  if (tid < R_RULES) {
    _Float16* brow = Bws + tid * K_PAD;
    float bias = 0.0f;
#pragma unroll
    for (int d = 0; d < D_FEAT; ++d) {
      // itertools.product(range(M), repeat=D): d=0 is the most significant digit
      int idx = (tid >> (D_FEAT - 1 - d)) & (M_MF - 1);
      float c = center[d * M_MF + idx];
      float s = sigma[d * M_MF + idx];
      float w = 1.0f / (s * s + EPS_F);
      brow[d]          = (_Float16)w;
      brow[D_FEAT + d] = (_Float16)(-2.0f * c * w);
      bias += c * c * w;
    }
#pragma unroll
    for (int k = 2 * D_FEAT; k < K_PAD; ++k) brow[k] = (_Float16)0.0f;
    biasws[tid] = bias;
  }
}

// ---------------------------------------------------------------------------
// Kernel 1: fused WMMA logits + row softmax.
// Block = 256 threads = 8 waves; owns 16 rows x 1024 cols.
// Each wave: 8 tiles of v_wmma_f32_16x16x32_f16, results staged in 64KB LDS,
// then per-row softmax (wave32 shfl reductions), coalesced store to HBM.
// ---------------------------------------------------------------------------
__global__ void __launch_bounds__(256)
fused_wmma_softmax_kernel(const _Float16* __restrict__ Aws,
                          const _Float16* __restrict__ Bws,
                          const float* __restrict__ biasws,
                          float* __restrict__ out) {
  __shared__ float zbuf[16 * R_RULES];   // 64 KB logit tile

  const int wave = threadIdx.x >> 5;     // 0..7
  const int lane = threadIdx.x & 31;     // wave32
  const int ln16 = lane & 15;
  const int koff = (lane < 16) ? 0 : 16; // per-lane K half (contiguous k = 16h + s, same for A and B)
  const int rowblk = blockIdx.x;         // 512 blocks of 16 rows

  // A fragment: row = rowblk*16 + ln16, K = koff..koff+15 contiguous halves.
  const int arow = rowblk * 16 + ln16;
  v16h afrag = *(const v16h*)(Aws + arow * K_PAD + koff);

  const float scale = -0.5f / (float)D_FEAT;   // HTSK mean + Gaussian -0.5

#pragma unroll
  for (int t = 0; t < 8; ++t) {
    const int ct  = wave * 8 + t;            // column tile 0..63
    const int col = ct * 16 + ln16;

    // B fragment: col-major rows of Bws, 16 contiguous halves per lane.
    v16h bfrag = *(const v16h*)(Bws + col * K_PAD + koff);

    v8f acc = {};
    acc = __builtin_amdgcn_wmma_f32_16x16x32_f16(
        false, afrag, false, bfrag, (short)0, acc, false, false);

    const float bv = biasws[col];
    const int row0 = (lane < 16) ? 0 : 8;    // C/D layout: VGPR j -> M=j / M=j+8
#pragma unroll
    for (int j = 0; j < 8; ++j) {
      zbuf[(row0 + j) * R_RULES + col] = scale * (acc[j] + bv);
    }
  }

  __syncthreads();

  // Softmax: wave w handles rows 2w and 2w+1; 32 lanes stride 1024 cols.
#pragma unroll
  for (int rr = 0; rr < 2; ++rr) {
    const int row  = wave * 2 + rr;
    float* zrow = zbuf + row * R_RULES;

    // 1) row max
    float m = -3.402823466e+38f;
#pragma unroll
    for (int j = 0; j < R_RULES / 32; ++j)
      m = fmaxf(m, zrow[lane + 32 * j]);
#pragma unroll
    for (int s = 16; s > 0; s >>= 1)
      m = fmaxf(m, __shfl_xor(m, s, 32));

    // 2) exp and sum (exp written back to LDS)
    float sum = 0.0f;
#pragma unroll
    for (int j = 0; j < R_RULES / 32; ++j) {
      float e = __expf(zrow[lane + 32 * j] - m);
      zrow[lane + 32 * j] = e;
      sum += e;
    }
#pragma unroll
    for (int s = 16; s > 0; s >>= 1)
      sum += __shfl_xor(sum, s, 32);

    // 3) normalize, coalesced global store
    const float inv = 1.0f / sum;
    float* orow = out + (size_t)(rowblk * 16 + row) * R_RULES;
#pragma unroll
    for (int j = 0; j < R_RULES / 32; ++j)
      orow[lane + 32 * j] = zrow[lane + 32 * j] * inv;
  }
}

// ---------------------------------------------------------------------------
// Host launcher
// ---------------------------------------------------------------------------
extern "C" void kernel_launch(void* const* d_in, const int* in_sizes, int n_in,
                              void* d_out, int out_size, void* d_ws, size_t ws_size,
                              hipStream_t stream) {
  const float* X      = (const float*)d_in[0];   // [N, 10]
  const float* center = (const float*)d_in[1];   // [10, 2]
  const float* sigma  = (const float*)d_in[2];   // [10, 2]
  float* out = (float*)d_out;                    // [N, 1024]

  const int N = in_sizes[0] / D_FEAT;            // 8192

  // Workspace layout (bytes): A | B | bias
  char* ws = (char*)d_ws;
  _Float16* Aws   = (_Float16*)ws;                                   // N*32 f16
  _Float16* Bws   = (_Float16*)(ws + (size_t)N * K_PAD * 2);         // 1024*32 f16
  float*    biasw = (float*)(ws + (size_t)N * K_PAD * 2
                                + (size_t)R_RULES * K_PAD * 2);      // 1024 f32

  // Kernel 0: build operands (covers max(N, R) threads)
  int prepThreads = (N > R_RULES) ? N : R_RULES;
  prep_kernel<<<(prepThreads + 255) / 256, 256, 0, stream>>>(
      X, center, sigma, Aws, Bws, biasw, N);

  // Kernel 1: one block per 16 rows
  fused_wmma_softmax_kernel<<<N / 16, 256, 0, stream>>>(Aws, Bws, biasw, out);
}
